// attentionb_36094905155585
// MI455X (gfx1250) — compile-verified, matrix operand order checked by reading, not run on hardware
//
#include <hip/hip_runtime.h>

typedef __attribute__((ext_vector_type(16))) __bf16 v16bf;
typedef __attribute__((ext_vector_type(8)))  __bf16 v8bf;
typedef __attribute__((ext_vector_type(8)))  float  v8f;

constexpr int   DIMS  = 1024;
constexpr int   NHEAD = 16;
constexpr int   HD    = 64;
constexpr int   BATCH = 2;
constexpr int   TSEQ  = 1024;
constexpr int   MROWS = BATCH * TSEQ;               // 2048
constexpr int   NEL   = BATCH * NHEAD * TSEQ * HD;  // 2,097,152
constexpr float SC_QK = 0.35355339059327373f;       // 64^-0.25
constexpr float INV_SQRT_HD = 0.125f;               // 1/sqrt(64)

struct FocusState {
  float diff_accum[4];
  int   cw;
  float temp;
  int   done;
  int   upd;
  int   keep_attn;
  float tscale;
};

// ---------------- WMMA helpers (CDNA5 16x16x32 bf16, f32 accum) ----------------

__device__ inline v8f wmma_bf16(v16bf a, v16bf b, v8f c) {
  return __builtin_amdgcn_wmma_f32_16x16x32_bf16(false, a, false, b, (short)0, c,
                                                 false, false);
}

// A-fragment (16xK slice, M=row) or B-fragment (N x K row-major weight slice).
// Lane l: row = l&15 ; K chunks at (half*8) and (half*8 + 16), 8 contiguous bf16 each.
__device__ inline v16bf frag_bf16(const __bf16* base, int ld, int lane, int k0) {
  int r    = lane & 15;
  int koff = (lane >> 4) << 3;
  const __bf16* p = base + r * ld + k0 + koff;
  v8bf a = *(const v8bf*)p;
  v8bf b = *(const v8bf*)(p + 16);
  v16bf f;
#pragma unroll
  for (int i = 0; i < 8; i++) { f[i] = a[i]; f[i + 8] = b[i]; }
  return f;
}

// Same fragment but converting from f32 source on the fly.
__device__ inline v16bf frag_from_f32(const float* base, int ld, int lane, int k0) {
  int r    = lane & 15;
  int koff = (lane >> 4) << 3;
  const float* p = base + r * ld + k0 + koff;
  v16bf f;
#pragma unroll
  for (int i = 0; i < 8; i++) f[i] = (__bf16)p[i];
  p += 16;
#pragma unroll
  for (int i = 0; i < 8; i++) f[8 + i] = (__bf16)p[i];
  return f;
}

// ---------------- misc small kernels ----------------

__global__ void init_state_kernel(FocusState* st, const float* temp_base) {
  if (threadIdx.x == 0 && blockIdx.x == 0) {
    for (int i = 0; i < 4; i++) st->diff_accum[i] = 0.0f;
    int cw0 = TSEQ;                    // max(min(T,2048),64)
    st->cw   = cw0;
    st->temp = *temp_base;
    st->done = 0;
    st->upd  = 0;
    st->keep_attn = 0;
    st->tscale = (st->temp > 0.0f) ? rsqrtf(st->temp) : 1.0f;
  }
}

__global__ void zero_f32_kernel(float* p, int n) {
  for (int i = blockIdx.x * blockDim.x + threadIdx.x; i < n; i += gridDim.x * blockDim.x)
    p[i] = 0.0f;
}

__global__ void cvt_f32_bf16_kernel(const float* __restrict__ in, __bf16* __restrict__ out, int n) {
  for (int i = blockIdx.x * blockDim.x + threadIdx.x; i < n; i += gridDim.x * blockDim.x)
    out[i] = (__bf16)in[i];
}

// Dual-output LayerNorm: xa = bf16(LN(x)*wa), xb = bf16(LN(x)*wb). One block per row.
__global__ void ln_dual_kernel(const float* __restrict__ x, const float* __restrict__ wa,
                               const float* __restrict__ wb, __bf16* __restrict__ xa,
                               __bf16* __restrict__ xb) {
  __shared__ float red[2][8];
  int row = blockIdx.x;
  const float* xr = x + (size_t)row * DIMS;
  float vals[4];
  float s = 0.0f, s2 = 0.0f;
#pragma unroll
  for (int i = 0; i < 4; i++) {
    float v = xr[threadIdx.x + i * 256];
    vals[i] = v; s += v; s2 += v * v;
  }
#pragma unroll
  for (int m = 1; m < 32; m <<= 1) { s += __shfl_xor(s, m, 32); s2 += __shfl_xor(s2, m, 32); }
  int wid = threadIdx.x >> 5, lane = threadIdx.x & 31;
  if (lane == 0) { red[0][wid] = s; red[1][wid] = s2; }
  __syncthreads();
  if (threadIdx.x == 0) {
    float a = 0.0f, b = 0.0f;
    for (int i = 0; i < 8; i++) { a += red[0][i]; b += red[1][i]; }
    red[0][0] = a; red[1][0] = b;
  }
  __syncthreads();
  s = red[0][0]; s2 = red[1][0];
  float mean = s * (1.0f / DIMS);
  float var  = s2 * (1.0f / DIMS) - mean * mean;
  float inv  = rsqrtf(var + 1e-5f);
#pragma unroll
  for (int i = 0; i < 4; i++) {
    int c = threadIdx.x + i * 256;
    float nv = (vals[i] - mean) * inv;
    xa[(size_t)row * DIMS + c] = (__bf16)(nv * wa[c]);
    xb[(size_t)row * DIMS + c] = (__bf16)(nv * wb[c]);
  }
}

// ---------------- big GEMM: C[M,N] = (A[M,K] @ W[N,K]^T + bias) * alpha ----------------
// 256 threads = 8 waves; wave computes 32x32 via 2x2 WMMA tiles; block tile 128x64.
__global__ void gemm_bf16_kernel(const __bf16* __restrict__ A, const __bf16* __restrict__ W,
                                 const float* __restrict__ bias, float alpha,
                                 float* __restrict__ C, int M, int N, int K) {
  int lane = threadIdx.x & 31;
  int wid  = threadIdx.x >> 5;
  int wm = wid >> 1, wn = wid & 1;
  int m0 = blockIdx.y * 128 + wm * 32;
  int n0 = blockIdx.x * 64 + wn * 32;
  v8f acc[2][2] = {};
  for (int k0 = 0; k0 < K; k0 += 32) {
    __builtin_prefetch(A + (size_t)m0 * K + k0 + 128, 0, 1);
    v16bf a0 = frag_bf16(A + (size_t)m0 * K, K, lane, k0);
    v16bf a1 = frag_bf16(A + (size_t)(m0 + 16) * K, K, lane, k0);
    v16bf b0 = frag_bf16(W + (size_t)n0 * K, K, lane, k0);
    v16bf b1 = frag_bf16(W + (size_t)(n0 + 16) * K, K, lane, k0);
    acc[0][0] = wmma_bf16(a0, b0, acc[0][0]);
    acc[0][1] = wmma_bf16(a0, b1, acc[0][1]);
    acc[1][0] = wmma_bf16(a1, b0, acc[1][0]);
    acc[1][1] = wmma_bf16(a1, b1, acc[1][1]);
  }
  int cm = (lane >> 4) << 3;   // row base within tile
  int cn = lane & 15;          // col within tile
#pragma unroll
  for (int i = 0; i < 2; i++)
#pragma unroll
    for (int j = 0; j < 2; j++) {
      int nn = n0 + j * 16 + cn;
      float bv = bias ? bias[nn] : 0.0f;
#pragma unroll
      for (int v = 0; v < 8; v++) {
        int mm = m0 + i * 16 + cm + v;
        C[(size_t)mm * N + nn] = (acc[i][j][v] + bv) * alpha;
      }
    }
}

// ---------------- per-head 64x64 projection helpers ----------------

// acc[4] += (A16x64_f32 @ W[64,64]^T) tile; then add bias per column.
__device__ inline void proj16x64(const float* Abase, const __bf16* Wb,
                                 const float* bias, int lane, v8f acc[4]) {
  for (int k0 = 0; k0 < HD; k0 += 32) {
    v16bf a = frag_from_f32(Abase, DIMS, lane, k0);
#pragma unroll
    for (int j = 0; j < 4; j++) {
      v16bf w = frag_bf16(Wb + j * 16 * HD, HD, lane, k0);
      acc[j] = wmma_bf16(a, w, acc[j]);
    }
  }
  int cn = lane & 15;
#pragma unroll
  for (int j = 0; j < 4; j++) {
    float bv = bias[j * 16 + cn];
#pragma unroll
    for (int v = 0; v < 8; v++) acc[j][v] += bv;
  }
}

// LayerNorm each of 16 rows (64 values spread over half-wave lanes x 4 tiles),
// scale by lnw and extraScale, store bf16 rows at outBase (ld = ldo).
__device__ inline void ln_rows_store(v8f acc[4], const float* lnw, float extraScale,
                                     int lane, __bf16* outBase, int ldo) {
  int cn = lane & 15;
  int half = lane >> 4;
#pragma unroll
  for (int v = 0; v < 8; v++) {
    float s = 0.0f, s2 = 0.0f;
#pragma unroll
    for (int j = 0; j < 4; j++) { float xv = acc[j][v]; s += xv; s2 += xv * xv; }
#pragma unroll
    for (int m = 1; m < 16; m <<= 1) { s += __shfl_xor(s, m, 32); s2 += __shfl_xor(s2, m, 32); }
    float mean = s * (1.0f / HD);
    float var  = s2 * (1.0f / HD) - mean * mean;
    float inv  = rsqrtf(var + 1e-5f) * extraScale;
    int mr = half * 8 + v;
#pragma unroll
    for (int j = 0; j < 4; j++) {
      int c = j * 16 + cn;
      outBase[mr * ldo + c] = (__bf16)((acc[j][v] - mean) * inv * lnw[c]);
    }
  }
}

// k/v head projections (loop-invariant): sk = bf16(LN(k@lkw^T+lkb)*lnd) [B,H,T,64],
// vpT = bf16(v@lvw^T+lvb) transposed [B,H,64,T].
__global__ void headproj_kv_kernel(const float* __restrict__ kbuf, const float* __restrict__ vbuf,
                                   const __bf16* __restrict__ lkw, const float* __restrict__ lkb,
                                   const __bf16* __restrict__ lvw, const float* __restrict__ lvb,
                                   const float* __restrict__ lnd_w,
                                   __bf16* __restrict__ sk, __bf16* __restrict__ vpT) {
  int lane = threadIdx.x & 31;
  int tile = blockIdx.x * (blockDim.x >> 5) + (threadIdx.x >> 5);
  int tt = tile % (TSEQ / 16);
  int bh = tile / (TSEQ / 16);
  int h = bh % NHEAD, b = bh / NHEAD;
  int t0 = tt * 16;

  const float* Kbase = kbuf + (size_t)(b * TSEQ + t0) * DIMS + h * HD;
  v8f acc[4] = {};
  proj16x64(Kbase, lkw, lkb, lane, acc);
  ln_rows_store(acc, lnd_w, 1.0f, lane,
                sk + (size_t)((b * NHEAD + h) * TSEQ + t0) * HD, HD);

  const float* Vbase = vbuf + (size_t)(b * TSEQ + t0) * DIMS + h * HD;
  v8f acc2[4] = {};
  proj16x64(Vbase, lvw, lvb, lane, acc2);
  int cn = lane & 15, half = lane >> 4;
#pragma unroll
  for (int j = 0; j < 4; j++)
#pragma unroll
    for (int v = 0; v < 8; v++) {
      int mr = half * 8 + v;
      int c  = j * 16 + cn;
      vpT[(size_t)((b * NHEAD + h) * HD + c) * TSEQ + (t0 + mr)] = (__bf16)acc2[j][v];
    }
}

// q head projection (per iteration): sq = bf16(LN(qcur@lqw^T+lqb)*lnc*tscale) [B,H,T,64]
__global__ void headproj_q_kernel(const float* __restrict__ qcur,
                                  const __bf16* __restrict__ lqw, const float* __restrict__ lqb,
                                  const float* __restrict__ lnc_w, const FocusState* __restrict__ st,
                                  __bf16* __restrict__ sq) {
  int lane = threadIdx.x & 31;
  int tile = blockIdx.x * (blockDim.x >> 5) + (threadIdx.x >> 5);
  int tt = tile % (TSEQ / 16);
  int bh = tile / (TSEQ / 16);
  int h = bh % NHEAD, b = bh / NHEAD;
  int t0 = tt * 16;
  float ts = st->tscale;
  const float* Qbase = qcur + (size_t)(b * TSEQ + t0) * DIMS + h * HD;
  v8f acc[4] = {};
  proj16x64(Qbase, lqw, lqb, lane, acc);
  ln_rows_store(acc, lnc_w, ts, lane,
                sq + (size_t)((b * NHEAD + h) * TSEQ + t0) * HD, HD);
}

// ---------------- attention: one wave per (b,h,16-row q strip) ----------------
// scores strip (16x1024 f32) + P (16x1024 bf16) in dynamic LDS (96 KB).
__global__ void attn_kernel(const __bf16* __restrict__ sq, const __bf16* __restrict__ skb,
                            const __bf16* __restrict__ vpT, const FocusState* __restrict__ st,
                            float* __restrict__ iter_out) {
  extern __shared__ char smem[];
  float*  Ssc = (float*)smem;                        // [16][1024]
  __bf16* Pb  = (__bf16*)(smem + 16 * 1024 * 4);     // [16][1024]
  int lane = threadIdx.x;                            // blockDim == 32
  int tile = blockIdx.x;
  int tt = tile % (TSEQ / 16);
  int bh = tile / (TSEQ / 16);
  int h = bh % NHEAD, b = bh / NHEAD;
  int t0 = tt * 16;
  int cw = st->cw;

  const __bf16* Aq = sq  + (size_t)((b * NHEAD + h) * TSEQ + t0) * HD;
  const __bf16* Bk = skb + (size_t)((b * NHEAD + h) * TSEQ) * HD;
  int cm = (lane >> 4) << 3, cn = lane & 15;

  v16bf a0 = frag_bf16(Aq, HD, lane, 0);
  v16bf a1 = frag_bf16(Aq, HD, lane, 32);
  for (int kt = 0; kt < TSEQ / 16; kt++) {
    v8f acc = {};
    acc = wmma_bf16(a0, frag_bf16(Bk + kt * 16 * HD, HD, lane, 0),  acc);
    acc = wmma_bf16(a1, frag_bf16(Bk + kt * 16 * HD, HD, lane, 32), acc);
#pragma unroll
    for (int v = 0; v < 8; v++)
      Ssc[(cm + v) * 1024 + kt * 16 + cn] = acc[v] * INV_SQRT_HD;
  }
  __syncthreads();

  // masked softmax: lane handles row (lane&15), columns [half*512, half*512+512)
  {
    int r  = lane & 15;
    int c0 = (lane >> 4) * 512;
    float mx = -1e30f;
    for (int c = c0; c < c0 + 512; c++)
      if (c < cw) mx = fmaxf(mx, Ssc[r * 1024 + c]);
    mx = fmaxf(mx, __shfl_xor(mx, 16, 32));
    float sum = 0.0f;
    for (int c = c0; c < c0 + 512; c++) {
      float e = (c < cw) ? __expf(Ssc[r * 1024 + c] - mx) : 0.0f;
      Ssc[r * 1024 + c] = e;
      sum += e;
    }
    sum += __shfl_xor(sum, 16, 32);
    float inv = 1.0f / sum;
    for (int c = c0; c < c0 + 512; c++)
      Pb[r * 1024 + c] = (__bf16)(Ssc[r * 1024 + c] * inv);
  }
  __syncthreads();

  // out = P @ Vp  (Vp^T stored row-major [64, T] -> contiguous B fragments)
  const __bf16* Vb = vpT + (size_t)((b * NHEAD + h) * HD) * TSEQ;
  v8f oacc[4] = {};
  for (int k0 = 0; k0 < TSEQ; k0 += 32) {
    v16bf pa = frag_bf16((const __bf16*)Pb, 1024, lane, k0);
#pragma unroll
    for (int j = 0; j < 4; j++) {
      v16bf vb = frag_bf16(Vb + (size_t)(j * 16) * TSEQ, TSEQ, lane, k0);
      oacc[j] = wmma_bf16(pa, vb, oacc[j]);
    }
  }
  // rows >= cw stay zero; write iter_out in [B,T,D] layout
#pragma unroll
  for (int j = 0; j < 4; j++)
#pragma unroll
    for (int v = 0; v < 8; v++) {
      int t = t0 + cm + v;
      float val = (t < cw) ? oacc[j][v] : 0.0f;
      iter_out[(size_t)(b * TSEQ + t) * DIMS + h * HD + j * 16 + cn] = val;
    }
}

// ---------------- convergence machinery ----------------

__global__ void reduce_diff_kernel(const float* __restrict__ a, const float* __restrict__ b,
                                   float* __restrict__ slot) {
  __shared__ float red[8];
  float s = 0.0f;
  for (int i = blockIdx.x * blockDim.x + threadIdx.x; i < NEL; i += gridDim.x * blockDim.x)
    s += fabsf(a[i] - b[i]);
#pragma unroll
  for (int m = 1; m < 32; m <<= 1) s += __shfl_xor(s, m, 32);
  int wid = threadIdx.x >> 5, lane = threadIdx.x & 31;
  if (lane == 0) red[wid] = s;
  __syncthreads();
  if (threadIdx.x == 0) {
    float t = 0.0f;
    for (int i = 0; i < 8; i++) t += red[i];
    atomicAdd(slot, t);
  }
}

__global__ void state_update_kernel(FocusState* st, int iter,
                                    const float* __restrict__ thrp,
                                    const float* __restrict__ facp) {
  if (threadIdx.x != 0 || blockIdx.x != 0) return;
  float diff = st->diff_accum[iter] * (1.0f / (float)NEL);
  float thr = *thrp, fac = *facp;
  int done = st->done;
  int converged = (diff < thr + fac * diff) && (iter > 0);
  int upd = (!done) && (!converged);
  st->keep_attn = done;
  st->upd = upd;
  st->done = done | converged;
  if (upd) {
    float cwf = (float)st->cw;
    int up = (int)(cwf * 1.1f);
    int dn = (int)(cwf * 0.9f);
    int nw = st->cw;
    if (diff > 0.05f) nw = up;
    else if (diff < 0.001f && st->cw > 64) nw = dn;
    if (nw < 64) nw = 64;
    if (nw > TSEQ) nw = TSEQ;
    st->cw = nw;
    st->temp += 0.005f;
  }
  st->tscale = (st->temp > 0.0f) ? rsqrtf(st->temp) : 1.0f;
}

__global__ void elemwise_update_kernel(const FocusState* __restrict__ st,
                                       const float* __restrict__ io, float* __restrict__ attnb,
                                       float* __restrict__ prevb, float* __restrict__ qcur) {
  int keep = st->keep_attn, upd = st->upd;
  for (int i = blockIdx.x * blockDim.x + threadIdx.x; i < NEL; i += gridDim.x * blockDim.x) {
    float v = io[i];
    if (!keep) attnb[i] = v;
    if (upd) { prevb[i] = v; qcur[i] += v; }
  }
}

// ---------------- launcher ----------------

extern "C" void kernel_launch(void* const* d_in, const int* in_sizes, int n_in,
                              void* d_out, int out_size, void* d_ws, size_t ws_size,
                              hipStream_t stream) {
  (void)in_sizes; (void)n_in; (void)out_size; (void)ws_size;
  const float* x    = (const float*)d_in[0];
  const float* Wq   = (const float*)d_in[1];
  const float* bq   = (const float*)d_in[2];
  const float* Wk   = (const float*)d_in[3];
  const float* Wv   = (const float*)d_in[4];
  const float* bv   = (const float*)d_in[5];
  const float* Wo   = (const float*)d_in[6];
  const float* bo   = (const float*)d_in[7];
  const float* lna  = (const float*)d_in[8];
  const float* lnb  = (const float*)d_in[9];
  const float* lnc  = (const float*)d_in[10];
  const float* lnd  = (const float*)d_in[11];
  const float* lqw  = (const float*)d_in[12];
  const float* lqb  = (const float*)d_in[13];
  const float* lkw  = (const float*)d_in[14];
  const float* lkb  = (const float*)d_in[15];
  const float* lvw  = (const float*)d_in[16];
  const float* lvb  = (const float*)d_in[17];
  const float* thr  = (const float*)d_in[18];
  const float* tmpb = (const float*)d_in[19];
  const float* fac  = (const float*)d_in[20];
  float* out = (float*)d_out;

  char* ws = (char*)d_ws;
  size_t off = 0;
  auto carve = [&](size_t bytes) {
    char* p = ws + off;
    off = (off + bytes + 255) & ~((size_t)255);
    return p;
  };
  FocusState* st = (FocusState*)carve(sizeof(FocusState));
  __bf16* wq_b  = (__bf16*)carve((size_t)DIMS * DIMS * 2);
  __bf16* wk_b  = (__bf16*)carve((size_t)DIMS * DIMS * 2);
  __bf16* wv_b  = (__bf16*)carve((size_t)DIMS * DIMS * 2);
  __bf16* wo_b  = (__bf16*)carve((size_t)DIMS * DIMS * 2);
  __bf16* lqw_b = (__bf16*)carve((size_t)HD * HD * 2);
  __bf16* lkw_b = (__bf16*)carve((size_t)HD * HD * 2);
  __bf16* lvw_b = (__bf16*)carve((size_t)HD * HD * 2);
  __bf16* xa_b  = (__bf16*)carve((size_t)MROWS * DIMS * 2);
  __bf16* xb_b  = (__bf16*)carve((size_t)MROWS * DIMS * 2);
  float*  qcur  = (float*)carve((size_t)MROWS * DIMS * 4);
  float*  kbuf  = (float*)carve((size_t)MROWS * DIMS * 4);
  float*  vbuf  = (float*)carve((size_t)MROWS * DIMS * 4);
  __bf16* sk_b  = (__bf16*)carve((size_t)NEL * 2);
  __bf16* vpT_b = (__bf16*)carve((size_t)NEL * 2);
  __bf16* sq_b  = (__bf16*)carve((size_t)NEL * 2);
  float*  iter_out = (float*)carve((size_t)NEL * 4);
  float*  prevb    = (float*)carve((size_t)NEL * 4);
  float*  attnb    = (float*)carve((size_t)NEL * 4);
  __bf16* attn_b   = (__bf16*)carve((size_t)NEL * 2);

  init_state_kernel<<<1, 1, 0, stream>>>(st, tmpb);
  zero_f32_kernel<<<2048, 256, 0, stream>>>(prevb, NEL);
  zero_f32_kernel<<<2048, 256, 0, stream>>>(attnb, NEL);

  cvt_f32_bf16_kernel<<<2048, 256, 0, stream>>>(Wq, wq_b, DIMS * DIMS);
  cvt_f32_bf16_kernel<<<2048, 256, 0, stream>>>(Wk, wk_b, DIMS * DIMS);
  cvt_f32_bf16_kernel<<<2048, 256, 0, stream>>>(Wv, wv_b, DIMS * DIMS);
  cvt_f32_bf16_kernel<<<2048, 256, 0, stream>>>(Wo, wo_b, DIMS * DIMS);
  cvt_f32_bf16_kernel<<<16, 256, 0, stream>>>(lqw, lqw_b, HD * HD);
  cvt_f32_bf16_kernel<<<16, 256, 0, stream>>>(lkw, lkw_b, HD * HD);
  cvt_f32_bf16_kernel<<<16, 256, 0, stream>>>(lvw, lvw_b, HD * HD);

  ln_dual_kernel<<<MROWS, 256, 0, stream>>>(x, lna, lnb, xa_b, xb_b);

  dim3 gg(DIMS / 64, MROWS / 128);
  gemm_bf16_kernel<<<gg, 256, 0, stream>>>(xa_b, wq_b, bq, SC_QK, qcur, MROWS, DIMS, DIMS);
  gemm_bf16_kernel<<<gg, 256, 0, stream>>>(xb_b, wk_b, nullptr, SC_QK, kbuf, MROWS, DIMS, DIMS);
  gemm_bf16_kernel<<<gg, 256, 0, stream>>>(xb_b, wv_b, bv, 1.0f, vbuf, MROWS, DIMS, DIMS);

  headproj_kv_kernel<<<512, 128, 0, stream>>>(kbuf, vbuf, lkw_b, lkb, lvw_b, lvb, lnd,
                                              sk_b, vpT_b);

  const int attn_lds = 16 * 1024 * 4 + 16 * 1024 * 2;  // 96 KB dynamic LDS
  for (int it = 0; it < 3; ++it) {
    headproj_q_kernel<<<512, 128, 0, stream>>>(qcur, lqw_b, lqb, lnc, st, sq_b);
    attn_kernel<<<BATCH * NHEAD * (TSEQ / 16), 32, attn_lds, stream>>>(sq_b, sk_b, vpT_b,
                                                                       st, iter_out);
    reduce_diff_kernel<<<1024, 256, 0, stream>>>(iter_out, prevb, &st->diff_accum[it]);
    state_update_kernel<<<1, 1, 0, stream>>>(st, it, thr, fac);
    elemwise_update_kernel<<<2048, 256, 0, stream>>>(st, iter_out, attnb, prevb, qcur);
  }

  cvt_f32_bf16_kernel<<<2048, 256, 0, stream>>>(attnb, attn_b, NEL);
  gemm_bf16_kernel<<<gg, 256, 0, stream>>>(attn_b, wo_b, bo, 1.0f, out, MROWS, DIMS, DIMS);
}
